// Transformer_Query_76579266887955
// MI455X (gfx1250) — compile-verified, matrix operand order checked by reading, not run on hardware
//
#include <hip/hip_runtime.h>
#include <hip/hip_bf16.h>

typedef _Float16 v16h __attribute__((ext_vector_type(16)));
typedef _Float16 v8h  __attribute__((ext_vector_type(8)));
typedef float    v8f  __attribute__((ext_vector_type(8)));

constexpr int BB   = 2048;
constexpr int SS   = 14;          // 2*QP + 2
constexpr int DD   = 200;
constexpr int DP   = 224;         // D padded to multiple of 32 for WMMA K
constexpr int NH   = 4;
constexpr int DH   = 50;
constexpr int FFD  = 800;         // already multiple of 32
constexpr int MM   = BB * SS;     // 28672 rows (multiple of 128)
constexpr int NENT = 50000;
constexpr int QPn  = 6;
constexpr float NEGF = -1e30f;

// ---------------------------------------------------------------------------
// f32[rows,K] -> f16[rows,KP] conversion with zero pad tail
// (optionally zero row 0, per reference ent/rel_embed.at[0].set(0))
// ---------------------------------------------------------------------------
__global__ __launch_bounds__(256) void convert_pad_kernel(const float* __restrict__ src,
                                                          _Float16* __restrict__ dst,
                                                          long long rows, int K, int KP,
                                                          int zero_first) {
  long long i = (long long)blockIdx.x * 256 + threadIdx.x;
  if (i < rows * KP) {
    long long r = i / KP;
    int k = (int)(i % KP);
    float v = 0.0f;
    if (k < K && !(zero_first && r == 0)) v = src[r * K + k];
    dst[i] = (_Float16)v;
  }
}

// Zero the pad tail [200,224) of an f16 activation buffer with row stride DP.
__global__ __launch_bounds__(256) void zero_tail_kernel(_Float16* __restrict__ buf,
                                                        long long rows) {
  long long i = (long long)blockIdx.x * 256 + threadIdx.x;
  long long n = rows * (DP - DD);
  if (i < n) {
    long long r = i / (DP - DD);
    int k = DD + (int)(i % (DP - DD));
    buf[r * DP + k] = (_Float16)0.0f;
  }
}

// ---------------------------------------------------------------------------
// Branch-free WMMA fragment loader. Row stride = Kp (padded, multiple of 32),
// so both 8-half chunks are unconditional 16-byte loads. Row index clamped
// (edge tiles compute garbage that is masked at store).
// lane%16 = row, lane/16 = g:  halves [kk+8g, kk+8g+8) and [kk+16+8g, +8).
// ---------------------------------------------------------------------------
__device__ __forceinline__ v16h load_frag(const _Float16* __restrict__ base,
                                          int row0, int nrows, int Kp, int kk, int lane) {
  int r = row0 + (lane & 15);
  r = r < nrows ? r : nrows - 1;
  int g = lane >> 4;
  const _Float16* rp = base + (size_t)r * (size_t)Kp;
  union { v16h v; v8h h8[2]; } f;
  f.h8[0] = *(const v8h*)(rp + kk + 8 * g);
  f.h8[1] = *(const v8h*)(rp + kk + 16 + 8 * g);
  return f.v;
}

// ---------------------------------------------------------------------------
// WMMA GEMM: C[M,N] = act(A[M,Kp] @ B[N,Kp]^T + bias[N])   (pad tail is zero)
// 256 threads = 8 waves, wave = 32x32 tile (4 wmma / 8 b128-loads per K step),
// block tile = 128(M) x 64(N).   EPI: 0 = none, 1 = relu, 2 = sigmoid
// ---------------------------------------------------------------------------
template <int EPI, typename OT>
__global__ __launch_bounds__(256) void gemm_wmma(const _Float16* __restrict__ A,
                                                 const _Float16* __restrict__ Bm,
                                                 const float* __restrict__ bias,
                                                 OT* __restrict__ C,
                                                 int M, int N, int Kp, int ldc) {
  int lane = threadIdx.x & 31;
  int wave = threadIdx.x >> 5;
  int wm = wave & 3;          // 4 waves along M
  int wn = wave >> 2;         // 2 waves along N
  int tileM = blockIdx.x * 128 + wm * 32;
  int tileN = blockIdx.y * 64 + wn * 32;

  union { v8f v; float f[8]; } acc[2][2];
#pragma unroll
  for (int i = 0; i < 2; ++i)
#pragma unroll
    for (int j = 0; j < 2; ++j)
      acc[i][j].v = (v8f){0.f, 0.f, 0.f, 0.f, 0.f, 0.f, 0.f, 0.f};

  for (int kk = 0; kk < Kp; kk += 32) {
    v16h a0 = load_frag(A, tileM, M, Kp, kk, lane);
    v16h a1 = load_frag(A, tileM + 16, M, Kp, kk, lane);
    v16h b0 = load_frag(Bm, tileN, N, Kp, kk, lane);
    v16h b1 = load_frag(Bm, tileN + 16, N, Kp, kk, lane);
    acc[0][0].v = __builtin_amdgcn_wmma_f32_16x16x32_f16(false, a0, false, b0, (short)0, acc[0][0].v, false, false);
    acc[0][1].v = __builtin_amdgcn_wmma_f32_16x16x32_f16(false, a0, false, b1, (short)0, acc[0][1].v, false, false);
    acc[1][0].v = __builtin_amdgcn_wmma_f32_16x16x32_f16(false, a1, false, b0, (short)0, acc[1][0].v, false, false);
    acc[1][1].v = __builtin_amdgcn_wmma_f32_16x16x32_f16(false, a1, false, b1, (short)0, acc[1][1].v, false, false);
  }

  int hv = lane >> 4;
#pragma unroll
  for (int j = 0; j < 2; ++j) {
    int n = tileN + j * 16 + (lane & 15);
    if (n >= N) continue;
    float bval = bias[n];
#pragma unroll
    for (int i = 0; i < 2; ++i) {
      int mbase = tileM + i * 16 + 8 * hv;
#pragma unroll
      for (int r = 0; r < 8; ++r) {
        int m = mbase + r;
        if (m < M) {
          float v = acc[i][j].f[r] + bval;
          if (EPI == 1) v = v > 0.0f ? v : 0.0f;
          if (EPI == 2) v = 1.0f / (1.0f + __expf(-v));
          C[(size_t)m * ldc + n] = (OT)v;
        }
      }
    }
  }
}

// ---------------------------------------------------------------------------
// Token builder: gathers, qualifier attention, mixture, pos embed, pad bias.
// One 256-thread block per batch element. XH written with stride DP.
// ---------------------------------------------------------------------------
__global__ __launch_bounds__(256) void build_tokens_kernel(
    const int* __restrict__ sub, const int* __restrict__ rel,
    const int* __restrict__ quals, const int* __restrict__ Nq,
    const float* __restrict__ ent, const float* __restrict__ relE,
    const float* __restrict__ w_sub, const float* __restrict__ w_rel,
    const float* __restrict__ w_method, const float* __restrict__ pos,
    float* __restrict__ X, _Float16* __restrict__ XH, float* __restrict__ padb) {
  int b = blockIdx.x;
  int tid = threadIdx.x;
  __shared__ float semb[DD], remb[DD];
  __shared__ float qe[QPn][DD], qr[QPn][DD];
  __shared__ float sa[DD], ra[DD];
  __shared__ float ssc[QPn], rsc[QPn];
  __shared__ float coef[2];

  int si = sub[b], ri = rel[b];
  for (int d = tid; d < DD; d += 256) {
    semb[d] = si ? ent[(size_t)si * DD + d] : 0.0f;
    remb[d] = ri ? relE[(size_t)ri * DD + d] : 0.0f;
  }
  for (int idx = tid; idx < QPn * DD; idx += 256) {
    int i = idx / DD, d = idx % DD;
    int ir = quals[b * (2 * QPn) + 2 * i];        // relation index
    int ie = quals[b * (2 * QPn) + 2 * i + 1];    // entity index
    qr[i][d] = ir ? relE[(size_t)ir * DD + d] : 0.0f;
    qe[i][d] = ie ? ent[(size_t)ie * DD + d] : 0.0f;
  }
  __syncthreads();

  if (tid < 12) {
    int i = tid % QPn;
    bool isSub = tid < QPn;
    const float* wv = isSub ? w_sub : w_rel;
    const float* te = isSub ? semb : remb;
    float dot = 0.0f;
    for (int d = 0; d < DD; ++d) dot += te[d] * wv[d];
    for (int d = 0; d < DD; ++d) dot += qr[i][d] * wv[DD + d];
    float lk = dot > 0.0f ? dot : 0.2f * dot;
    bool valid = Nq[b * QPn + i] > 0;
    float e = valid ? __expf(-lk) : 0.0f;   // exp(NEG) -> 0
    (isSub ? ssc : rsc)[i] = e;
  }
  __syncthreads();
  if (tid < 2) {
    float* arr = (tid == 0) ? ssc : rsc;
    float tot = 0.0f;
    for (int i = 0; i < QPn; ++i) tot += arr[i];
    if (tot == 0.0f) tot = 1.0f;
    for (int i = 0; i < QPn; ++i) arr[i] /= tot;
  }
  __syncthreads();
  for (int d = tid; d < DD; d += 256) {
    float s = 0.0f, r = 0.0f;
    for (int i = 0; i < QPn; ++i) { s += qe[i][d] * ssc[i]; r += qe[i][d] * rsc[i]; }
    sa[d] = s; ra[d] = r;
  }
  __syncthreads();
  if (tid < 2) {
    const float* src = (tid == 0) ? sa : ra;
    float dot = 0.0f;
    for (int d = 0; d < DD; ++d) dot += src[d] * w_method[d];
    float lk = dot > 0.0f ? dot : 0.2f * dot;
    coef[tid] = __expf(-lk);
  }
  __syncthreads();
  float es = coef[0], er = coef[1], tot = es + er;
  float ws = es / tot, wr = er / tot;

  size_t rb  = (size_t)b * SS * DD;   // f32 X rows
  size_t rbh = (size_t)b * SS * DP;   // f16 XH rows (padded)
  for (int d = tid; d < DD; d += 256) {
    float emb = ws * sa[d] + wr * ra[d];
    float v0 = (0.5f * semb[d] + 0.5f * emb) + pos[0 * DD + d];
    float v1 = (0.5f * remb[d] + 0.5f * emb) + pos[1 * DD + d];
    X[rb + 0 * DD + d] = v0; XH[rbh + 0 * DP + d] = (_Float16)v0;
    X[rb + 1 * DD + d] = v1; XH[rbh + 1 * DP + d] = (_Float16)v1;
  }
  for (int idx = tid; idx < QPn * DD; idx += 256) {
    int i = idx / DD, d = idx % DD;
    float ve = qe[i][d] + pos[(2 + 2 * i) * DD + d];
    float vr = qr[i][d] + pos[(3 + 2 * i) * DD + d];
    X[rb + (2 + 2 * i) * DD + d] = ve; XH[rbh + (2 + 2 * i) * DP + d] = (_Float16)ve;
    X[rb + (3 + 2 * i) * DD + d] = vr; XH[rbh + (3 + 2 * i) * DP + d] = (_Float16)vr;
  }
  if (tid < SS) {
    float pb = 0.0f;
    if (tid >= 2) { int i = (tid - 2) >> 1; pb = (Nq[b * QPn + i] > 0) ? 0.0f : NEGF; }
    padb[b * SS + tid] = pb;
  }
}

// ---------------------------------------------------------------------------
// Multi-head attention over S=14 tokens. One 64-thread block per batch.
// q/k/v staged in LDS, thread = (head, query-pos), output written f16 (stride DP).
// ---------------------------------------------------------------------------
__global__ __launch_bounds__(64) void attn_kernel(const float* __restrict__ QKV,
                                                  const float* __restrict__ padb,
                                                  _Float16* __restrict__ OH) {
  int b = blockIdx.x;
  __shared__ float q[SS * DD], k[SS * DD], v[SS * DD];
  const float* base = QKV + (size_t)b * SS * (3 * DD);
  for (int idx = threadIdx.x; idx < SS * DD; idx += 64) {
    int s = idx / DD, d = idx % DD;
    q[idx] = base[s * (3 * DD) + d];
    k[idx] = base[s * (3 * DD) + DD + d];
    v[idx] = base[s * (3 * DD) + 2 * DD + d];
  }
  __syncthreads();
  int t = threadIdx.x;
  if (t < NH * SS) {
    int h = t / SS, s = t % SS;
    float scale = rsqrtf((float)DH);
    float sc[SS];
    float mx = -3.0e38f;
#pragma unroll
    for (int j = 0; j < SS; ++j) {
      float dot = 0.0f;
      for (int d = 0; d < DH; ++d) dot += q[s * DD + h * DH + d] * k[j * DD + h * DH + d];
      sc[j] = dot * scale + padb[b * SS + j];
      mx = fmaxf(mx, sc[j]);
    }
    float tot = 0.0f;
#pragma unroll
    for (int j = 0; j < SS; ++j) { sc[j] = __expf(sc[j] - mx); tot += sc[j]; }
    float inv = 1.0f / tot;
    for (int d = 0; d < DH; ++d) {
      float o = 0.0f;
#pragma unroll
      for (int j = 0; j < SS; ++j) o += sc[j] * v[j * DD + h * DH + d];
      OH[((size_t)b * SS + s) * DP + h * DH + d] = (_Float16)(o * inv);
    }
  }
}

// ---------------------------------------------------------------------------
// Residual + LayerNorm over D=200; writes f32 (residual stream) + f16 copy.
// One 64-thread block per row.
// ---------------------------------------------------------------------------
__global__ __launch_bounds__(64) void resid_ln_kernel(float* __restrict__ X,
                                                      const float* __restrict__ R,
                                                      const float* __restrict__ g,
                                                      const float* __restrict__ bta,
                                                      _Float16* __restrict__ XH) {
  int row = blockIdx.x;
  int tid = threadIdx.x;
  __shared__ float vbuf[DD];
  __shared__ float red[64];
  __shared__ float mu_s, rstd_s;
  size_t base = (size_t)row * DD;
  size_t baseh = (size_t)row * DP;
  float s = 0.0f;
  for (int d = tid; d < DD; d += 64) { float t = X[base + d] + R[base + d]; vbuf[d] = t; s += t; }
  red[tid] = s;
  __syncthreads();
  if (tid == 0) { float tot = 0.0f; for (int i = 0; i < 64; ++i) tot += red[i]; mu_s = tot / DD; }
  __syncthreads();
  float mu = mu_s;
  float s2 = 0.0f;
  for (int d = tid; d < DD; d += 64) { float t = vbuf[d] - mu; s2 += t * t; }
  red[tid] = s2;
  __syncthreads();
  if (tid == 0) { float tot = 0.0f; for (int i = 0; i < 64; ++i) tot += red[i]; rstd_s = rsqrtf(tot / DD + 1e-5f); }
  __syncthreads();
  float rs = rstd_s;
  for (int d = tid; d < DD; d += 64) {
    float y = (vbuf[d] - mu) * rs * g[d] + bta[d];
    X[base + d] = y;
    XH[baseh + d] = (_Float16)y;
  }
}

// ---------------------------------------------------------------------------
// Mean over S, batchnorm stats/apply
// ---------------------------------------------------------------------------
__global__ __launch_bounds__(256) void pool_mean_kernel(const float* __restrict__ X,
                                                        float* __restrict__ pooled) {
  int idx = blockIdx.x * 256 + threadIdx.x;
  if (idx < BB * DD) {
    int b = idx / DD, d = idx % DD;
    float s = 0.0f;
    for (int j = 0; j < SS; ++j) s += X[((size_t)b * SS + j) * DD + d];
    pooled[idx] = s * (1.0f / SS);
  }
}

__global__ __launch_bounds__(256) void bn_stats_kernel(const float* __restrict__ pooled,
                                                       float* __restrict__ mu,
                                                       float* __restrict__ rvar) {
  int d = blockIdx.x;
  int tid = threadIdx.x;
  __shared__ float r1[256], r2[256];
  float s = 0.0f, s2 = 0.0f;
  for (int b = tid; b < BB; b += 256) {
    float v = pooled[(size_t)b * DD + d];
    s += v; s2 += v * v;
  }
  r1[tid] = s; r2[tid] = s2;
  __syncthreads();
  if (tid == 0) {
    float a = 0.0f, c = 0.0f;
    for (int i = 0; i < 256; ++i) { a += r1[i]; c += r2[i]; }
    float m = a / BB;
    mu[d] = m;
    rvar[d] = rsqrtf(c / BB - m * m + 1e-5f);
  }
}

__global__ __launch_bounds__(256) void bn_apply_kernel(const float* __restrict__ pooled,
                                                       const float* __restrict__ mu,
                                                       const float* __restrict__ rvar,
                                                       const float* __restrict__ gamma,
                                                       const float* __restrict__ beta,
                                                       _Float16* __restrict__ YH) {
  int idx = blockIdx.x * 256 + threadIdx.x;
  if (idx < BB * DD) {
    int b = idx / DD, d = idx % DD;
    float v = (pooled[idx] - mu[d]) * rvar[d] * gamma[d] + beta[d];
    v = v > 0.0f ? v : 0.0f;
    YH[(size_t)b * DP + d] = (_Float16)v;
  }
}

// ---------------------------------------------------------------------------
// Host-side orchestration
// ---------------------------------------------------------------------------
extern "C" void kernel_launch(void* const* d_in, const int* in_sizes, int n_in,
                              void* d_out, int out_size, void* d_ws, size_t ws_size,
                              hipStream_t stream) {
  (void)in_sizes; (void)n_in; (void)out_size; (void)ws_size;
  const int*   sub      = (const int*)d_in[0];
  const int*   rel      = (const int*)d_in[1];
  const int*   quals    = (const int*)d_in[2];
  const int*   Nq       = (const int*)d_in[3];
  const float* ent      = (const float*)d_in[4];
  const float* relE     = (const float*)d_in[5];
  const float* w_sub    = (const float*)d_in[6];
  const float* w_rel    = (const float*)d_in[7];
  const float* w_method = (const float*)d_in[8];
  const float* pos      = (const float*)d_in[9];
  const float* out_bias = (const float*)d_in[10];
  const float* bn_g     = (const float*)d_in[11];
  const float* bn_b     = (const float*)d_in[12];
  const float* in_w     = (const float*)d_in[13];
  const float* in_b     = (const float*)d_in[14];
  const float* out_w    = (const float*)d_in[15];
  const float* out_b    = (const float*)d_in[16];
  const float* ln1g     = (const float*)d_in[17];
  const float* ln1b     = (const float*)d_in[18];
  const float* ln2g     = (const float*)d_in[19];
  const float* ln2b     = (const float*)d_in[20];
  const float* ff1w     = (const float*)d_in[21];
  const float* ff1b     = (const float*)d_in[22];
  const float* ff2w     = (const float*)d_in[23];
  const float* ff2b     = (const float*)d_in[24];

  char* w = (char*)d_ws;
  size_t off = 0;
  auto take = [&](size_t bytes) {
    char* p = w + off;
    off += (bytes + 255) & ~(size_t)255;
    return p;
  };

  _Float16* ENT16  = (_Float16*)take((size_t)NENT * DP * 2);
  _Float16* WIN16  = (_Float16*)take((size_t)2 * 3 * DD * DP * 2);
  _Float16* WOUT16 = (_Float16*)take((size_t)2 * DD * DP * 2);
  _Float16* WF116  = (_Float16*)take((size_t)2 * FFD * DP * 2);
  _Float16* WF216  = (_Float16*)take((size_t)2 * DD * FFD * 2);     // K=800, no pad
  float*    X      = (float*)take((size_t)MM * DD * 4);
  _Float16* XH     = (_Float16*)take((size_t)MM * DP * 2);
  float*    QKV    = (float*)take((size_t)MM * 3 * DD * 4);          // reused as HH (f16 MM x 800)
  _Float16* HHb    = (_Float16*)QKV;
  _Float16* OH     = (_Float16*)take((size_t)MM * DP * 2);
  float*    TMP    = (float*)take((size_t)MM * DD * 4);
  float*    PADB   = (float*)take((size_t)MM * 4);
  float*    POOL   = (float*)take((size_t)BB * DD * 4);
  float*    MU     = (float*)take(DD * 4);
  float*    RV     = (float*)take(DD * 4);
  _Float16* YH     = (_Float16*)take((size_t)BB * DP * 2);

  auto cg = [](long long n) { return (unsigned)((n + 255) / 256); };

  // Embedding / weight conversion to padded f16 (row 0 of ent zeroed per reference)
  convert_pad_kernel<<<cg((long long)NENT * DP), 256, 0, stream>>>(ent, ENT16, NENT, DD, DP, 1);
  convert_pad_kernel<<<cg(2LL * 3 * DD * DP), 256, 0, stream>>>(in_w, WIN16, 2LL * 3 * DD, DD, DP, 0);
  convert_pad_kernel<<<cg(2LL * DD * DP), 256, 0, stream>>>(out_w, WOUT16, 2LL * DD, DD, DP, 0);
  convert_pad_kernel<<<cg(2LL * FFD * DP), 256, 0, stream>>>(ff1w, WF116, 2LL * FFD, DD, DP, 0);
  convert_pad_kernel<<<cg(2LL * DD * FFD), 256, 0, stream>>>(ff2w, WF216, 2LL * DD, FFD, FFD, 0);

  // Zero pad tails of padded f16 activation buffers (writers only touch d < 200)
  zero_tail_kernel<<<cg((long long)MM * (DP - DD)), 256, 0, stream>>>(XH, MM);
  zero_tail_kernel<<<cg((long long)MM * (DP - DD)), 256, 0, stream>>>(OH, MM);
  zero_tail_kernel<<<cg((long long)BB * (DP - DD)), 256, 0, stream>>>(YH, BB);

  // Build token sequence (gathers + qualifier attention + pos embed + pad bias)
  build_tokens_kernel<<<BB, 256, 0, stream>>>(sub, rel, quals, Nq, ent, relE,
                                              w_sub, w_rel, w_method, pos, X, XH, PADB);

  // Transformer layers
  for (int l = 0; l < 2; ++l) {
    const _Float16* Wqkv = WIN16 + (size_t)l * 3 * DD * DP;
    const _Float16* Wout = WOUT16 + (size_t)l * DD * DP;
    const _Float16* Wf1  = WF116 + (size_t)l * FFD * DP;
    const _Float16* Wf2  = WF216 + (size_t)l * DD * FFD;

    gemm_wmma<0, float><<<dim3(MM / 128, (3 * DD + 63) / 64), 256, 0, stream>>>(
        XH, Wqkv, in_b + (size_t)l * 3 * DD, QKV, MM, 3 * DD, DP, 3 * DD);

    attn_kernel<<<BB, 64, 0, stream>>>(QKV, PADB, OH);

    gemm_wmma<0, float><<<dim3(MM / 128, (DD + 63) / 64), 256, 0, stream>>>(
        OH, Wout, out_b + (size_t)l * DD, TMP, MM, DD, DP, DD);

    resid_ln_kernel<<<MM, 64, 0, stream>>>(X, TMP, ln1g + (size_t)l * DD, ln1b + (size_t)l * DD, XH);

    gemm_wmma<1, _Float16><<<dim3(MM / 128, (FFD + 63) / 64), 256, 0, stream>>>(
        XH, Wf1, ff1b + (size_t)l * FFD, HHb, MM, FFD, DP, FFD);

    gemm_wmma<0, float><<<dim3(MM / 128, (DD + 63) / 64), 256, 0, stream>>>(
        HHb, Wf2, ff2b + (size_t)l * DD, TMP, MM, DD, FFD, DD);

    resid_ln_kernel<<<MM, 64, 0, stream>>>(X, TMP, ln2g + (size_t)l * DD, ln2b + (size_t)l * DD, XH);
  }

  // Pool over tokens, batchnorm, relu
  pool_mean_kernel<<<cg((long long)BB * DD), 256, 0, stream>>>(X, POOL);
  bn_stats_kernel<<<DD, 256, 0, stream>>>(POOL, MU, RV);
  bn_apply_kernel<<<cg((long long)BB * DD), 256, 0, stream>>>(POOL, MU, RV, bn_g, bn_b, YH);

  // Final scoring GEMM: sigmoid(Y @ ent^T + bias) -> d_out (2048 x 50000 f32)
  gemm_wmma<2, float><<<dim3(BB / 128, (NENT + 63) / 64), 256, 0, stream>>>(
      YH, ENT16, out_bias, (float*)d_out, BB, NENT, DP, NENT);
}
